// MultiHeadAttention_1537598292043
// MI455X (gfx1250) — matrix operand order repaired and checked by reading, not compiled, not run on hardware
//
#include <hip/hip_runtime.h>
#include <hip/hip_bf16.h>

// ---------------------------------------------------------------------------
// MI455X (gfx1250) multi-head attention, bf16 WMMA (v_wmma_f32_16x16x32_bf16)
// ---------------------------------------------------------------------------

typedef __attribute__((ext_vector_type(16))) __bf16 v16bf;
typedef __attribute__((ext_vector_type(8)))  __bf16 bf16x8;
typedef __attribute__((ext_vector_type(8)))  float  v8f;

#define B_   8
#define S_   1024
#define D_   512
#define H_   8
#define DK_  64

// ---- WMMA helpers ----------------------------------------------------------

__device__ __forceinline__ v8f wmma_bf16(v16bf a, v16bf b, v8f c) {
    // D = A(16x32 bf16) * B(32x16 bf16) + C(16x16 f32)
    return __builtin_amdgcn_wmma_f32_16x16x32_bf16(
        /*neg_a=*/false, a, /*neg_b=*/false, b,
        /*c_mod=*/(short)0, c, /*reuse_a=*/false, /*reuse_b=*/false);
}

// A fragment (16x32, bf16 source, row-major, row already selected):
// lanes 0-15 : K = {0..15} (ab=0)
// lanes 16-31: K = {16..31} (ab=8)
__device__ __forceinline__ v16bf load_a_bf16(const __bf16* rowk0, int ab) {
    bf16x8 lo = *(const bf16x8*)(rowk0 + 2 * ab);
    bf16x8 hi = *(const bf16x8*)(rowk0 + 2 * ab + 8);
    return __builtin_shufflevector(lo, hi,
        0,1,2,3,4,5,6,7,8,9,10,11,12,13,14,15);
}

// A fragment from f32 source, converted to bf16 in registers
__device__ __forceinline__ v16bf load_a_f32(const float* rowk0, int ab) {
    const float4* p0 = (const float4*)(rowk0 + 2 * ab);
    const float4* p1 = (const float4*)(rowk0 + 2 * ab + 8);
    float4 c0 = p0[0], c1 = p0[1];
    float4 c2 = p1[0], c3 = p1[1];
    v16bf a;
    a[0]  = (__bf16)c0.x; a[1]  = (__bf16)c0.y; a[2]  = (__bf16)c0.z; a[3]  = (__bf16)c0.w;
    a[4]  = (__bf16)c1.x; a[5]  = (__bf16)c1.y; a[6]  = (__bf16)c1.z; a[7]  = (__bf16)c1.w;
    a[8]  = (__bf16)c2.x; a[9]  = (__bf16)c2.y; a[10] = (__bf16)c2.z; a[11] = (__bf16)c2.w;
    a[12] = (__bf16)c3.x; a[13] = (__bf16)c3.y; a[14] = (__bf16)c3.z; a[15] = (__bf16)c3.w;
    return a;
}

// B fragment (32x16): column n fixed per lane, 16 contiguous K values
// (lanes 0-15: K=k0..k0+15, lanes 16-31: K=k0+16..k0+31 -> caller adds kb)
__device__ __forceinline__ v16bf load_b_bf16(const __bf16* colk) {
    const bf16x8* p = (const bf16x8*)colk;
    bf16x8 lo = p[0], hi = p[1];
    return __builtin_shufflevector(lo, hi,
        0,1,2,3,4,5,6,7,8,9,10,11,12,13,14,15);
}

// ---- weight transpose + f32->bf16 convert: wt[n][k] = w[k][n] --------------

__global__ void wconv_kernel(const float* __restrict__ w, __bf16* __restrict__ wt) {
    int k = blockIdx.x * 16 + threadIdx.x;
    int n = blockIdx.y * 16 + threadIdx.y;
    wt[(size_t)n * D_ + k] = (__bf16)w[(size_t)k * D_ + n];
}

// ---- projection GEMM: C[16x64] tile per wave -------------------------------
// MODE 0: Q/K  -> head layout  [b,h,s,dk] bf16, scaled
// MODE 1: V    -> head layout transposed [b,h,dk,s] bf16
// MODE 2: Out  -> f32 d_out (+bias +residual), A is bf16
template<int MODE>
__global__ void proj_gemm_kernel(const float* __restrict__ Af32,
                                 const __bf16* __restrict__ Abf,
                                 const __bf16* __restrict__ Bt,
                                 const float* __restrict__ bias,
                                 const float* __restrict__ residual,
                                 __bf16* __restrict__ outBf,
                                 float* __restrict__ outF32,
                                 float scale) {
    const int tid   = threadIdx.x;
    const int lane  = tid & 31;
    const int gw    = blockIdx.x * 4 + (tid >> 5);   // 4096 waves total
    const int tileM = (gw >> 3) * 16;                // 512 row tiles
    const int tileN = (gw & 7) * 64;                 // 8 col groups
    const int rlow  = lane & 15;
    const int ab    = (lane < 16) ? 0 : 8;
    const int kb    = (lane < 16) ? 0 : 16;
    const int mBase = (lane < 16) ? 0 : 8;
    const int arow  = tileM + rlow;

    v8f zero = {};
    v8f acc[4];
#pragma unroll
    for (int s = 0; s < 4; ++s) acc[s] = zero;

    for (int k0 = 0; k0 < D_; k0 += 32) {
        v16bf a;
        if constexpr (MODE == 2) a = load_a_bf16(Abf + (size_t)arow * D_ + k0, ab);
        else                     a = load_a_f32 (Af32 + (size_t)arow * D_ + k0, ab);
#pragma unroll
        for (int sub = 0; sub < 4; ++sub) {
            const __bf16* brow = Bt + (size_t)(tileN + sub * 16 + rlow) * D_ + k0 + kb;
            v16bf bfrag = load_b_bf16(brow);
            acc[sub] = wmma_bf16(a, bfrag, acc[sub]);
        }
    }

#pragma unroll
    for (int sub = 0; sub < 4; ++sub) {
#pragma unroll
        for (int r = 0; r < 8; ++r) {
            float val = acc[sub][r];
            int t = tileM + mBase + r;               // global row (b*S+s)
            int d = tileN + sub * 16 + rlow;         // global col
            if constexpr (MODE == 0) {
                int b = t >> 10, s = t & 1023, h = d >> 6, dk = d & 63;
                float v = (val + bias[d]) * scale;
                outBf[((size_t)(b * H_ + h) * S_ + s) * DK_ + dk] = (__bf16)v;
            } else if constexpr (MODE == 1) {
                int b = t >> 10, s = t & 1023, h = d >> 6, dk = d & 63;
                float v = val + bias[d];
                outBf[((size_t)(b * H_ + h) * DK_ + dk) * S_ + s] = (__bf16)v;
            } else {
                size_t idx = (size_t)t * D_ + d;
                outF32[idx] = val + bias[d] + residual[idx];
            }
        }
    }
}

// ---- attention scores: 16x16 tile per wave, fused epilogue -----------------
// adj/dist/mask (96 MB total) are streamed once per head: non-temporal loads keep the
// 192 MB L2 free for the heavily reused qh/kh/vht bf16 operands.

__global__ void scores_kernel(const __bf16* __restrict__ qh,
                              const __bf16* __restrict__ kh,
                              const float* __restrict__ adj,
                              const float* __restrict__ dist,
                              const int*   __restrict__ mask,
                              float* __restrict__ attn) {
    const int tid  = threadIdx.x;
    const int lane = tid & 31;
    const int gw   = blockIdx.x * 4 + (tid >> 5);    // 262144 waves
    const int kt   = gw & 63;
    const int qt   = (gw >> 6) & 63;
    const int h    = (gw >> 12) & 7;
    const int b    = gw >> 15;
    const int bh   = b * H_ + h;
    const int rlow = lane & 15;
    const int ab   = (lane < 16) ? 0 : 8;
    const int kb   = (lane < 16) ? 0 : 16;
    const int mBase = (lane < 16) ? 0 : 8;

    const __bf16* abase = qh + ((size_t)bh * S_ + (qt * 16 + rlow)) * DK_;
    const __bf16* bbase = kh + ((size_t)bh * S_ + (kt * 16 + rlow)) * DK_;

    v8f acc = {};
#pragma unroll
    for (int k0 = 0; k0 < DK_; k0 += 32) {
        v16bf a  = load_a_bf16(abase + k0, ab);
        v16bf bf = load_b_bf16(bbase + k0 + kb);
        acc = wmma_bf16(a, bf, acc);
    }

    const float E1 = 2.718281828459045f;
#pragma unroll
    for (int r = 0; r < 8; ++r) {
        int q = qt * 16 + mBase + r;
        int k = kt * 16 + rlow;
        size_t eidx = ((size_t)b * S_ + q) * S_ + k;     // (B,S,S) / (B,1,S,S)
        float dval = __builtin_nontemporal_load(dist + eidx);
        float aval = __builtin_nontemporal_load(adj  + eidx);
        int   mval = __builtin_nontemporal_load(mask + eidx);
        float resc = (1.0f + E1) / (1.0f + __expf(1.0f - dval));
        float val  = fmaxf(acc[r], 0.0f) * resc + aval;
        if (mval == 0) val = -1e9f;
        attn[((size_t)bh * S_ + q) * S_ + k] = val;
    }
}

// ---- softmax over 1024 keys: one wave per row, in place --------------------

__global__ void softmax_kernel(float* __restrict__ attn) {
    const int lane = threadIdx.x & 31;
    const int row  = blockIdx.x * 8 + (threadIdx.x >> 5);  // 65536 rows
    float4* p = (float4*)(attn + (size_t)row * S_);

    float4 v[8];
    float m = -3.4e38f;
#pragma unroll
    for (int i = 0; i < 8; ++i) {
        v[i] = p[i * 32 + lane];
        m = fmaxf(m, fmaxf(fmaxf(v[i].x, v[i].y), fmaxf(v[i].z, v[i].w)));
    }
#pragma unroll
    for (int off = 16; off > 0; off >>= 1) m = fmaxf(m, __shfl_xor(m, off, 32));

    float sum = 0.0f;
#pragma unroll
    for (int i = 0; i < 8; ++i) {
        v[i].x = __expf(v[i].x - m); v[i].y = __expf(v[i].y - m);
        v[i].z = __expf(v[i].z - m); v[i].w = __expf(v[i].w - m);
        sum += v[i].x + v[i].y + v[i].z + v[i].w;
    }
#pragma unroll
    for (int off = 16; off > 0; off >>= 1) sum += __shfl_xor(sum, off, 32);

    float inv = 1.0f / sum;
#pragma unroll
    for (int i = 0; i < 8; ++i) {
        float4 o = v[i];
        o.x *= inv; o.y *= inv; o.z *= inv; o.w *= inv;
        p[i * 32 + lane] = o;
    }
}

// ---- attn @ V: 16x64 tile per wave (f32 attn converted to bf16 on load) ----

__global__ void attn_v_kernel(const float* __restrict__ attn,
                              const __bf16* __restrict__ vht,
                              __bf16* __restrict__ attnOut) {
    const int tid  = threadIdx.x;
    const int lane = tid & 31;
    const int gw   = blockIdx.x * 4 + (tid >> 5);    // 4096 waves
    const int qt   = gw & 63;
    const int h    = (gw >> 6) & 7;
    const int b    = gw >> 9;
    const int bh   = b * H_ + h;
    const int rlow = lane & 15;
    const int ab   = (lane < 16) ? 0 : 8;
    const int kb   = (lane < 16) ? 0 : 16;
    const int mBase = (lane < 16) ? 0 : 8;

    const float*  abase = attn + ((size_t)bh * S_ + (qt * 16 + rlow)) * S_;
    const __bf16* vbase = vht + (size_t)bh * DK_ * S_;

    v8f zero = {};
    v8f acc[4];
#pragma unroll
    for (int s = 0; s < 4; ++s) acc[s] = zero;

    for (int k0 = 0; k0 < S_; k0 += 32) {
        v16bf a = load_a_f32(abase + k0, ab);
#pragma unroll
        for (int sub = 0; sub < 4; ++sub) {
            const __bf16* brow = vbase + (size_t)(sub * 16 + rlow) * S_ + k0 + kb;
            v16bf bf = load_b_bf16(brow);
            acc[sub] = wmma_bf16(a, bf, acc[sub]);
        }
    }

#pragma unroll
    for (int sub = 0; sub < 4; ++sub) {
#pragma unroll
        for (int r = 0; r < 8; ++r) {
            int q = qt * 16 + mBase + r;
            int d = h * DK_ + sub * 16 + rlow;       // concat heads
            attnOut[((size_t)b * S_ + q) * D_ + d] = (__bf16)acc[sub][r];
        }
    }
}

// ---- layernorm over D=512, one wave per row, in place ----------------------

__global__ void layernorm_kernel(float* __restrict__ out,
                                 const float* __restrict__ g,
                                 const float* __restrict__ beta) {
    const int lane = threadIdx.x & 31;
    const int row  = blockIdx.x * 8 + (threadIdx.x >> 5);  // 8192 rows
    float4* p = (float4*)(out + (size_t)row * D_);
    const float4* g4 = (const float4*)g;
    const float4* b4 = (const float4*)beta;

    float4 v[4];
    float sum = 0.0f, sq = 0.0f;
#pragma unroll
    for (int i = 0; i < 4; ++i) {
        v[i] = p[i * 32 + lane];
        sum += v[i].x + v[i].y + v[i].z + v[i].w;
        sq  += v[i].x * v[i].x + v[i].y * v[i].y + v[i].z * v[i].z + v[i].w * v[i].w;
    }
#pragma unroll
    for (int off = 16; off > 0; off >>= 1) {
        sum += __shfl_xor(sum, off, 32);
        sq  += __shfl_xor(sq,  off, 32);
    }
    float mean = sum * (1.0f / (float)D_);
    float var  = sq  * (1.0f / (float)D_) - mean * mean;
    float inv  = rsqrtf(var + 1e-6f);

#pragma unroll
    for (int i = 0; i < 4; ++i) {
        int idx = i * 32 + lane;
        float4 gg = g4[idx], bb = b4[idx], o;
        o.x = (v[i].x - mean) * inv * gg.x + bb.x;
        o.y = (v[i].y - mean) * inv * gg.y + bb.y;
        o.z = (v[i].z - mean) * inv * gg.z + bb.z;
        o.w = (v[i].w - mean) * inv * gg.w + bb.w;
        p[idx] = o;
    }
}

// ---------------------------------------------------------------------------

extern "C" void kernel_launch(void* const* d_in, const int* in_sizes, int n_in,
                              void* d_out, int out_size, void* d_ws, size_t ws_size,
                              hipStream_t stream) {
    const float* q    = (const float*)d_in[0];
    const float* k    = (const float*)d_in[1];
    const float* v    = (const float*)d_in[2];
    const int*   mask = (const int*)  d_in[3];
    const float* adj  = (const float*)d_in[4];
    const float* dist = (const float*)d_in[5];
    const float* wq   = (const float*)d_in[6];
    const float* bq   = (const float*)d_in[7];
    const float* wk   = (const float*)d_in[8];
    const float* bk   = (const float*)d_in[9];
    const float* wv   = (const float*)d_in[10];
    const float* bv   = (const float*)d_in[11];
    const float* wo   = (const float*)d_in[12];
    const float* bo   = (const float*)d_in[13];
    const float* ln_g = (const float*)d_in[14];
    const float* ln_b = (const float*)d_in[15];

    float* outPtr  = (float*)d_out;                          // [B,S,D]
    float* attnPtr = outPtr + (size_t)B_ * S_ * D_;          // [B,H,S,S]

    // workspace layout (all 16B aligned)
    char* ws = (char*)d_ws;
    size_t off = 0;
    __bf16* wqt  = (__bf16*)(ws + off); off += (size_t)D_ * D_ * 2;
    __bf16* wkt  = (__bf16*)(ws + off); off += (size_t)D_ * D_ * 2;
    __bf16* wvt  = (__bf16*)(ws + off); off += (size_t)D_ * D_ * 2;
    __bf16* wot  = (__bf16*)(ws + off); off += (size_t)D_ * D_ * 2;
    __bf16* qh   = (__bf16*)(ws + off); off += (size_t)B_ * H_ * S_ * DK_ * 2;
    __bf16* kh   = (__bf16*)(ws + off); off += (size_t)B_ * H_ * S_ * DK_ * 2;
    __bf16* vht  = (__bf16*)(ws + off); off += (size_t)B_ * H_ * S_ * DK_ * 2;
    __bf16* aout = (__bf16*)(ws + off); off += (size_t)B_ * S_ * D_ * 2;
    (void)off; (void)in_sizes; (void)n_in; (void)out_size; (void)ws_size;

    dim3 wb(16, 16), wg(32, 32);
    wconv_kernel<<<wg, wb, 0, stream>>>(wq, wqt);
    wconv_kernel<<<wg, wb, 0, stream>>>(wk, wkt);
    wconv_kernel<<<wg, wb, 0, stream>>>(wv, wvt);
    wconv_kernel<<<wg, wb, 0, stream>>>(wo, wot);

    // projections: 4096 waves, 4 waves per block
    proj_gemm_kernel<0><<<1024, 128, 0, stream>>>(q, nullptr, wqt, bq, nullptr, qh,  nullptr, 0.125f); // 1/sqrt(DK)
    proj_gemm_kernel<0><<<1024, 128, 0, stream>>>(k, nullptr, wkt, bk, nullptr, kh,  nullptr, 1.0f);
    proj_gemm_kernel<1><<<1024, 128, 0, stream>>>(v, nullptr, wvt, bv, nullptr, vht, nullptr, 1.0f);

    // scores (pre-softmax, fused relu/rescale/adj/mask) -> d_out attn region
    scores_kernel<<<65536, 128, 0, stream>>>(qh, kh, adj, dist, mask, attnPtr);
    // softmax in place
    softmax_kernel<<<8192, 256, 0, stream>>>(attnPtr);
    // attn @ V -> concat-head bf16
    attn_v_kernel<<<1024, 128, 0, stream>>>(attnPtr, vht, aout);
    // output projection + bias + residual -> f32 out region
    proj_gemm_kernel<2><<<1024, 128, 0, stream>>>(nullptr, aout, wot, bo, q, nullptr, outPtr, 1.0f);
    // layernorm in place
    layernorm_kernel<<<1024, 256, 0, stream>>>(outPtr, ln_g, ln_b);
}
